// FullAttention_39290360824429
// MI455X (gfx1250) — matrix operand order repaired and checked by reading, not compiled, not run on hardware
//
#include <hip/hip_runtime.h>
#include <hip/hip_bf16.h>

// ---------------------------------------------------------------------------
// MHA forward for MI455X (gfx1250, wave32, WMMA).
// Precision: bf16 operands, fp32 WMMA accumulation (v_wmma_f32_16x16x32_bf16).
// GEMM kernels use 2x2 register blocking (32x32 per wave) so each K-step is
// 4 WMMAs on 2 A-frags + 2 B-frags (2x data reuse vs. naive tiling).
// ---------------------------------------------------------------------------

typedef __attribute__((ext_vector_type(16))) __bf16 v16bf;
typedef __attribute__((ext_vector_type(8)))  __bf16 v8bf;
typedef __attribute__((ext_vector_type(8)))  float  v8f;

constexpr int CB   = 2;      // batch
constexpr int CT   = 2048;   // seq len
constexpr int CC   = 1024;   // embed
constexpr int CH   = 16;     // heads
constexpr int CD   = 64;     // head dim
constexpr float SCALE = 0.125f;  // 1/sqrt(64)

// ---- fragment helpers (CDNA5 ISA layouts, wave32) ---------------------
__device__ __forceinline__ v16bf frag_combine(v8bf lo, v8bf hi8) {
  v16bf r;
#pragma unroll
  for (int i = 0; i < 8; ++i) { r[i] = lo[i]; r[i + 8] = hi8[i]; }
  return r;
}
// A-matrix 16x32 bf16: lane<16 holds K {0..7,16..23}; lane>=16 holds {8..15,24..31}
__device__ __forceinline__ v16bf load_fragA(const __bf16* rowk, int hi) {
  const __bf16* p = rowk + (hi ? 8 : 0);
  return frag_combine(*(const v8bf*)p, *(const v8bf*)(p + 16));
}
// B-matrix 32x16 bf16: lane<16 holds K {0..7,16..23}; lane>=16 holds {8..15,24..31}
__device__ __forceinline__ v16bf load_fragB(const __bf16* rowk, int hi) {
  const __bf16* p = rowk + (hi ? 8 : 0);
  return frag_combine(*(const v8bf*)p, *(const v8bf*)(p + 16));
}
__device__ __forceinline__ v8f wmma_bf16(v16bf a, v16bf b, v8f c) {
  return __builtin_amdgcn_wmma_f32_16x16x32_bf16(false, a, false, b,
                                                 (short)0, c, false, false);
}

// ---------------------------------------------------------------------------
// K0: fp32 -> bf16 conversion of x and the four weight matrices
// ---------------------------------------------------------------------------
__global__ void k_convert(const float* __restrict__ x,
                          const float* __restrict__ Wq, const float* __restrict__ Wk,
                          const float* __restrict__ Wv, const float* __restrict__ Wp,
                          __bf16* __restrict__ xb,
                          __bf16* __restrict__ wqb, __bf16* __restrict__ wkb,
                          __bf16* __restrict__ wvb, __bf16* __restrict__ wpb) {
  const int XN = CB * CT * CC;           // 4,194,304
  const int WN = CC * CC;                // 1,048,576
  int i = blockIdx.x * blockDim.x + threadIdx.x;
  if (i < XN) {
    xb[i] = (__bf16)x[i];
  } else {
    int j = i - XN;
    int w = j >> 20;                     // which weight
    int o = j & (WN - 1);
    if      (w == 0) wqb[o] = (__bf16)Wq[o];
    else if (w == 1) wkb[o] = (__bf16)Wk[o];
    else if (w == 2) wvb[o] = (__bf16)Wv[o];
    else             wpb[o] = (__bf16)Wp[o];
  }
}

// ---------------------------------------------------------------------------
// K1: QKV projections.  out = x @ W^T + b.  One wave per 32x32 output block
// (2x2 WMMA tiles, 2x A/B reuse).  Q,K stored [B,H,T,D]; V stored [B,H,D,T].
// ---------------------------------------------------------------------------
__global__ void __launch_bounds__(256)
k_qkv(const __bf16* __restrict__ xb,
      const __bf16* __restrict__ wqb, const __bf16* __restrict__ wkb,
      const __bf16* __restrict__ wvb,
      const float* __restrict__ bq, const float* __restrict__ bk,
      const float* __restrict__ bv,
      __bf16* __restrict__ Qb, __bf16* __restrict__ Kb, __bf16* __restrict__ Vt) {
  int lane = threadIdx.x & 31;
  int wid  = blockIdx.x * 8 + (threadIdx.x >> 5);
  int mat  = wid >> 12;                  // 0:Q 1:K 2:V   (4096 blocks per matrix)
  int r    = wid & 4095;
  int m0   = (r >> 5) << 5;              // 32-row block base in [0,4096)
  int n0   = (r & 31) << 5;              // 32-col block base in [0,1024)
  int l15  = lane & 15, hi = lane >> 4;

  const __bf16* W    = (mat == 0) ? wqb : (mat == 1) ? wkb : wvb;
  const float*  bias = (mat == 0) ? bq  : (mat == 1) ? bk  : bv;

  const __bf16* arow0 = xb + (size_t)(m0 + l15) * CC;
  const __bf16* arow1 = arow0 + (size_t)16 * CC;
  const __bf16* brow0 = W  + (size_t)(n0 + l15) * CC;
  const __bf16* brow1 = brow0 + (size_t)16 * CC;

  v8f acc[2][2] = {};
#pragma unroll 2
  for (int c0 = 0; c0 < CC; c0 += 32) {
    v16bf a0 = load_fragA(arow0 + c0, hi);
    v16bf a1 = load_fragA(arow1 + c0, hi);
    v16bf b0 = load_fragB(brow0 + c0, hi);
    v16bf b1 = load_fragB(brow1 + c0, hi);
    acc[0][0] = wmma_bf16(a0, b0, acc[0][0]);
    acc[0][1] = wmma_bf16(a0, b1, acc[0][1]);
    acc[1][0] = wmma_bf16(a1, b0, acc[1][0]);
    acc[1][1] = wmma_bf16(a1, b1, acc[1][1]);
  }

#pragma unroll
  for (int mi = 0; mi < 2; ++mi) {
#pragma unroll
    for (int ni = 0; ni < 2; ++ni) {
      int msb  = m0 + mi * 16;
      int nsb  = n0 + ni * 16;
      float bval = bias[nsb + l15];
      int bidx = msb >> 11;              // batch
      int t0   = msb & (CT - 1);
      int h    = nsb >> 6;
      int d    = (nsb & 63) + l15;
      if (mat == 2) {
#pragma unroll
        for (int v = 0; v < 8; ++v) {
          int t = t0 + v + 8 * hi;
          Vt[(((size_t)(bidx * CH + h) * CD) + d) * CT + t] =
              (__bf16)(acc[mi][ni][v] + bval);
        }
      } else {
        __bf16* dst = (mat == 0) ? Qb : Kb;
#pragma unroll
        for (int v = 0; v < 8; ++v) {
          int t = t0 + v + 8 * hi;
          dst[(((size_t)(bidx * CH + h) * CT) + t) * CD + d] =
              (__bf16)(acc[mi][ni][v] + bval);
        }
      }
    }
  }
}

// ---------------------------------------------------------------------------
// K2: softmax row statistics (online max / sum-exp) per (b,h,t).
// One wave per (b,h, t-tile of 16 rows), streaming over all s-tiles.
// ---------------------------------------------------------------------------
__global__ void __launch_bounds__(256)
k_stats(const __bf16* __restrict__ Qb, const __bf16* __restrict__ Kb,
        const unsigned char* __restrict__ mask,
        float* __restrict__ mstat, float* __restrict__ lstat) {
  int lane = threadIdx.x & 31;
  int wid  = blockIdx.x * 8 + (threadIdx.x >> 5);   // 4096 waves
  int bh   = wid >> 7;                              // b*16+h
  int t0   = (wid & 127) << 4;
  int b    = bh >> 4;
  int l15  = lane & 15, hi = lane >> 4;

  const __bf16* qbase = Qb + (size_t)bh * CT * CD;
  const __bf16* kbase = Kb + (size_t)bh * CT * CD;
  const __bf16* qrow  = qbase + (size_t)(t0 + l15) * CD;
  v16bf qa0 = load_fragA(qrow, hi);
  v16bf qa1 = load_fragA(qrow + 32, hi);

  float mrun[8], lrun[8];
#pragma unroll
  for (int v = 0; v < 8; ++v) { mrun[v] = -1e30f; lrun[v] = 0.f; }

  const unsigned char* mbase = mask + (size_t)b * CT * CT;

  for (int s0 = 0; s0 < CT; s0 += 16) {
    const __bf16* krow = kbase + (size_t)(s0 + l15) * CD;
    __builtin_prefetch(krow + 16 * CD);             // next s-tile
    v16bf kb0 = load_fragB(krow, hi);
    v16bf kb1 = load_fragB(krow + 32, hi);
    v8f c = {};
    c = wmma_bf16(qa0, kb0, c);
    c = wmma_bf16(qa1, kb1, c);
#pragma unroll
    for (int v = 0; v < 8; ++v) {
      int t = t0 + v + 8 * hi;
      float sc = c[v] * SCALE;
      if (mbase[(size_t)t * CT + s0 + l15]) sc = -1e30f;
      float tm = sc;
#pragma unroll
      for (int o = 1; o < 16; o <<= 1) tm = fmaxf(tm, __shfl_xor(tm, o, 32));
      float nm = fmaxf(mrun[v], tm);
      float e = __expf(sc - nm);
#pragma unroll
      for (int o = 1; o < 16; o <<= 1) e += __shfl_xor(e, o, 32);
      lrun[v] = lrun[v] * __expf(mrun[v] - nm) + e;
      mrun[v] = nm;
    }
  }
  if (l15 == 0) {
#pragma unroll
    for (int v = 0; v < 8; ++v) {
      int t = t0 + v + 8 * hi;
      mstat[(size_t)bh * CT + t] = mrun[v];
      lstat[(size_t)bh * CT + t] = lrun[v];
    }
  }
}

// ---------------------------------------------------------------------------
// K3a: att_mean[b,t,s] = (1/H) * sum_h softmax(qk)[b,h,t,s]
// One wave per (b, t-tile, s-tile); heads looped in registers -> no atomics.
// ---------------------------------------------------------------------------
__global__ void __launch_bounds__(256)
k_attmean(const __bf16* __restrict__ Qb, const __bf16* __restrict__ Kb,
          const unsigned char* __restrict__ mask,
          const float* __restrict__ mstat, const float* __restrict__ lstat,
          float* __restrict__ attm) {
  int lane = threadIdx.x & 31;
  int wid  = blockIdx.x * 8 + (threadIdx.x >> 5);   // 32768 waves
  int b    = wid >> 14;
  int r    = wid & 16383;
  int t0   = (r >> 7) << 4;
  int s0   = (r & 127) << 4;
  int l15  = lane & 15, hi = lane >> 4;

  bool mk[8];
#pragma unroll
  for (int v = 0; v < 8; ++v)
    mk[v] = mask[((size_t)b * CT + t0 + v + 8 * hi) * CT + s0 + l15] != 0;

  float acc[8];
#pragma unroll
  for (int v = 0; v < 8; ++v) acc[v] = 0.f;

  for (int h = 0; h < CH; ++h) {
    int bh = b * CH + h;
    const __bf16* qrow = Qb + (size_t)bh * CT * CD + (size_t)(t0 + l15) * CD;
    const __bf16* krow = Kb + (size_t)bh * CT * CD + (size_t)(s0 + l15) * CD;
    v16bf qa0 = load_fragA(qrow, hi);
    v16bf qa1 = load_fragA(qrow + 32, hi);
    v16bf kb0 = load_fragB(krow, hi);
    v16bf kb1 = load_fragB(krow + 32, hi);
    v8f c = {};
    c = wmma_bf16(qa0, kb0, c);
    c = wmma_bf16(qa1, kb1, c);
#pragma unroll
    for (int v = 0; v < 8; ++v) {
      int t = t0 + v + 8 * hi;
      float m = mstat[(size_t)bh * CT + t];
      float l = lstat[(size_t)bh * CT + t];
      float p = mk[v] ? 0.f : __expf(c[v] * SCALE - m) / l;
      acc[v] += p;
    }
  }
#pragma unroll
  for (int v = 0; v < 8; ++v)
    attm[((size_t)b * CT + t0 + v + 8 * hi) * CT + s0 + l15] = acc[v] * (1.f / CH);
}

// ---------------------------------------------------------------------------
// K3b: y = P @ V (flash-style).  One wave per (b,h, t-tile of 16 rows).
// P tiles are re-fragmented A-side through a stride-72 (conflict-free) LDS tile.
// y written bf16 into [B,T,C] layout for the final projection.
// ---------------------------------------------------------------------------
__global__ void __launch_bounds__(256)
k_av(const __bf16* __restrict__ Qb, const __bf16* __restrict__ Kb,
     const __bf16* __restrict__ Vt, const unsigned char* __restrict__ mask,
     const float* __restrict__ mstat, const float* __restrict__ lstat,
     __bf16* __restrict__ yb) {
  __shared__ __align__(16) __bf16 lds[8][16 * 72];   // per-wave 16x32 P tile, stride 72
  int lane = threadIdx.x & 31;
  int wv   = threadIdx.x >> 5;
  int wid  = blockIdx.x * 8 + wv;                    // 4096 waves
  int bh   = wid >> 7;
  int t0   = (wid & 127) << 4;
  int b    = bh >> 4;
  int h    = bh & 15;
  int l15  = lane & 15, hi = lane >> 4;

  const __bf16* qrow  = Qb + (size_t)bh * CT * CD + (size_t)(t0 + l15) * CD;
  const __bf16* kbase = Kb + (size_t)bh * CT * CD;
  const __bf16* vbase = Vt + (size_t)bh * CD * CT;
  v16bf qa0 = load_fragA(qrow, hi);
  v16bf qa1 = load_fragA(qrow + 32, hi);

  float mrow[8], invl[8];
#pragma unroll
  for (int v = 0; v < 8; ++v) {
    int t = t0 + v + 8 * hi;
    mrow[v] = mstat[(size_t)bh * CT + t];
    invl[v] = 1.f / lstat[(size_t)bh * CT + t];
  }

  v8f acc[4] = {};
  __bf16* pl = &lds[wv][0];
  const unsigned char* mbase = mask + (size_t)b * CT * CT;

  for (int s0 = 0; s0 < CT; s0 += 32) {
#pragma unroll
    for (int j = 0; j < 2; ++j) {
      int sb = s0 + j * 16;
      const __bf16* krow = kbase + (size_t)(sb + l15) * CD;
      __builtin_prefetch(krow + 32 * CD);
      v8f c = {};
      c = wmma_bf16(qa0, load_fragB(krow, hi), c);
      c = wmma_bf16(qa1, load_fragB(krow + 32, hi), c);
#pragma unroll
      for (int v = 0; v < 8; ++v) {
        int t = t0 + v + 8 * hi;
        bool mk = mbase[(size_t)t * CT + sb + l15] != 0;
        float p = mk ? 0.f : __expf(c[v] * SCALE - mrow[v]) * invl[v];
        pl[(v + 8 * hi) * 72 + j * 16 + l15] = (__bf16)p;
      }
    }
    __syncthreads();                                 // uniform across all 8 waves
    v16bf pa = load_fragA(pl + l15 * 72, hi);        // P as A-fragment (16x32)
#pragma unroll
    for (int dt = 0; dt < 4; ++dt) {
      const __bf16* vrow = vbase + (size_t)(dt * 16 + l15) * CT + s0;
      acc[dt] = wmma_bf16(pa, load_fragB(vrow, hi), acc[dt]);
    }
    __syncthreads();
  }
#pragma unroll
  for (int dt = 0; dt < 4; ++dt)
#pragma unroll
    for (int v = 0; v < 8; ++v) {
      int t = t0 + v + 8 * hi;
      yb[((size_t)b * CT + t) * CC + h * CD + dt * 16 + l15] = (__bf16)acc[dt][v];
    }
}

// ---------------------------------------------------------------------------
// K4: output projection  out = y @ Wp^T + bp   (fp32 result)
// One wave per 32x32 output block (2x2 WMMA tiles, 2x A/B reuse).
// ---------------------------------------------------------------------------
__global__ void __launch_bounds__(256)
k_proj(const __bf16* __restrict__ yb, const __bf16* __restrict__ wpb,
       const float* __restrict__ bp, float* __restrict__ yout) {
  int lane = threadIdx.x & 31;
  int wid  = blockIdx.x * 8 + (threadIdx.x >> 5);   // 4096 wave-blocks
  int m0   = (wid >> 5) << 5;
  int n0   = (wid & 31) << 5;
  int l15  = lane & 15, hi = lane >> 4;

  const __bf16* arow0 = yb  + (size_t)(m0 + l15) * CC;
  const __bf16* arow1 = arow0 + (size_t)16 * CC;
  const __bf16* brow0 = wpb + (size_t)(n0 + l15) * CC;
  const __bf16* brow1 = brow0 + (size_t)16 * CC;

  v8f acc[2][2] = {};
#pragma unroll 2
  for (int c0 = 0; c0 < CC; c0 += 32) {
    v16bf a0 = load_fragA(arow0 + c0, hi);
    v16bf a1 = load_fragA(arow1 + c0, hi);
    v16bf b0 = load_fragB(brow0 + c0, hi);
    v16bf b1 = load_fragB(brow1 + c0, hi);
    acc[0][0] = wmma_bf16(a0, b0, acc[0][0]);
    acc[0][1] = wmma_bf16(a0, b1, acc[0][1]);
    acc[1][0] = wmma_bf16(a1, b0, acc[1][0]);
    acc[1][1] = wmma_bf16(a1, b1, acc[1][1]);
  }

#pragma unroll
  for (int mi = 0; mi < 2; ++mi) {
#pragma unroll
    for (int ni = 0; ni < 2; ++ni) {
      float bval = bp[n0 + ni * 16 + l15];
#pragma unroll
      for (int v = 0; v < 8; ++v)
        yout[(size_t)(m0 + mi * 16 + v + 8 * hi) * CC + n0 + ni * 16 + l15] =
            acc[mi][ni][v] + bval;
    }
  }
}

// ---------------------------------------------------------------------------
extern "C" void kernel_launch(void* const* d_in, const int* in_sizes, int n_in,
                              void* d_out, int out_size, void* d_ws, size_t ws_size,
                              hipStream_t stream) {
  const float* x  = (const float*)d_in[0];
  // d_in[1] = encoder_output (unused by reference forward)
  const unsigned char* mask = (const unsigned char*)d_in[2];
  const float* Wq = (const float*)d_in[3];
  const float* bq = (const float*)d_in[4];
  const float* Wk = (const float*)d_in[5];
  const float* bk = (const float*)d_in[6];
  const float* Wv = (const float*)d_in[7];
  const float* bv = (const float*)d_in[8];
  const float* Wp = (const float*)d_in[9];
  const float* bp = (const float*)d_in[10];

  float* yout = (float*)d_out;
  float* attm = yout + (size_t)CB * CT * CC;

  // workspace layout (bytes); total ~48.5 MB
  char* w = (char*)d_ws;
  __bf16* xb  = (__bf16*)(w + 0);                     //  8 MB  x bf16
  __bf16* wqb = (__bf16*)(w + 8388608);               //  2 MB
  __bf16* wkb = (__bf16*)(w + 8388608 + 2097152);     //  2 MB
  __bf16* wvb = (__bf16*)(w + 8388608 + 2 * 2097152); //  2 MB
  __bf16* wpb = (__bf16*)(w + 8388608 + 3 * 2097152); //  2 MB
  __bf16* Qb  = (__bf16*)(w + 16777216);              //  8 MB  [B,H,T,D]
  __bf16* Kb  = (__bf16*)(w + 25165824);              //  8 MB  [B,H,T,D]
  __bf16* Vt  = (__bf16*)(w + 33554432);              //  8 MB  [B,H,D,T]
  __bf16* yb  = (__bf16*)(w + 41943040);              //  8 MB  [B,T,C]
  float*  mst = (float*)(w + 50331648);               // 256 KB [B,H,T]
  float*  lst = (float*)(w + 50331648 + 262144);      // 256 KB [B,H,T]

  // K0: conversions (4M x-elems + 4 * 1M weight-elems)
  k_convert<<<(8388608 + 255) / 256, 256, 0, stream>>>(
      x, Wq, Wk, Wv, Wp, xb, wqb, wkb, wvb, wpb);

  // K1: QKV projections: 3 * 128 * 32 = 12288 wave-blocks, 8 waves/block
  k_qkv<<<1536, 256, 0, stream>>>(xb, wqb, wkb, wvb, bq, bk, bv, Qb, Kb, Vt);

  // K2: softmax stats: 2*16*128 = 4096 waves
  k_stats<<<512, 256, 0, stream>>>(Qb, Kb, mask, mst, lst);

  // K3a: att_mean: 2*128*128 = 32768 waves
  k_attmean<<<4096, 256, 0, stream>>>(Qb, Kb, mask, mst, lst, attm);

  // K3b: y = P@V: 4096 waves
  k_av<<<512, 256, 0, stream>>>(Qb, Kb, Vt, mask, mst, lst, yb);

  // K4: output projection: 128 * 32 = 4096 wave-blocks
  k_proj<<<512, 256, 0, stream>>>(yb, wpb, bp, yout);
}
